// Local_rep_layer_49143015801020
// MI455X (gfx1250) — compile-verified, matrix operand order checked by reading, not run on hardware
//
#include <hip/hip_runtime.h>
#include <hip/hip_bf16.h>

// ---- problem constants ----
#define NB 32    // docs
#define NE 48    // entities/doc
#define NM 96    // mentions/doc
#define ND 256   // hidden
#define NH 8     // heads
#define NPAIR (NE*NE)          // 2304
#define PAIR_TILES (NPAIR/16)  // 144

typedef __attribute__((ext_vector_type(16))) __bf16 v16bf;
typedef __attribute__((ext_vector_type(8)))  float  v8f;

union BFrag { uint4 u[2]; v16bf v; };

__device__ __forceinline__ unsigned short f2bf(float f) {
    unsigned int u = __float_as_uint(f);
    u += 0x7FFFu + ((u >> 16) & 1u);   // round-to-nearest-even
    return (unsigned short)(u >> 16);
}

// ---------------- prep kernels ----------------

// W [4][256][256] f32 row-major -> WT [4][256(N)][256(K)] bf16
__global__ void k_cast_transpose_w(unsigned short* __restrict__ wt,
                                   const float* __restrict__ w) {
    int t = blockIdx.x * 256 + threadIdx.x;           // 4*256*256 threads
    int s = t >> 16, rem = t & 65535;
    int k = rem >> 8, n = rem & 255;
    wt[((size_t)s * 256 + n) * 256 + k] = f2bf(w[((size_t)s * 256 + k) * 256 + n]);
}

__global__ void k_cast_bf(unsigned short* __restrict__ dst,
                          const float* __restrict__ src, int n) {
    int t = blockIdx.x * 256 + threadIdx.x;
    if (t < n) dst[t] = f2bf(src[t]);
}

// men_sen_bf[r][d] = bf16(sentences[info[r,4]][d]),  r < NB*NM
__global__ void k_gather_sen(unsigned short* __restrict__ dst,
                             const float* __restrict__ sentences,
                             const int* __restrict__ info) {
    int t = blockIdx.x * 256 + threadIdx.x;           // NB*NM*256 threads
    int r = t >> 8, d = t & 255;
    int sid = info[r * 6 + 4];
    dst[t] = f2bf(sentences[(size_t)sid * 256 + d]);
}

// per (doc,entity): list of mentions owned by that entity
__global__ void k_build_lists(int* __restrict__ cnt, int* __restrict__ idx,
                              const int* __restrict__ info) {
    int be = blockIdx.x * 256 + threadIdx.x;          // NB*NE threads
    if (be >= NB * NE) return;
    int b = be / NE;
    int c = 0;
    int* lst = idx + (size_t)be * NM;
    for (int m = 0; m < NM; ++m)
        if (info[((size_t)b * NM + m) * 6] == be) lst[c++] = m;
    cnt[be] = c;
}

// ---------------- generic bf16 WMMA GEMM: out[rows x 256] = A @ W + bias ----
// A: bf16 [rows][256] row-major; WT: bf16 [256(N)][256(K)]; out f32.
// grid.x = rows/16, block = 256 (8 waves); wave w -> col tiles 2w, 2w+1.
__global__ void k_proj_gemm(const unsigned short* __restrict__ A,
                            const unsigned short* __restrict__ WT,
                            const float* __restrict__ bias,
                            float* __restrict__ out) {
    __shared__ unsigned short As[16][264];
    int row0 = blockIdx.x * 16;
    int t = threadIdx.x;
    for (int c = t; c < 512; c += 256) {              // stage 16x256 bf16 tile
        int r = c >> 5, c8 = c & 31;
        *(uint4*)&As[r][c8 * 8] = *(const uint4*)&A[(size_t)(row0 + r) * 256 + c8 * 8];
    }
    __syncthreads();
    int lane = t & 31, wv = t >> 5;
    int arow = lane & 15, koff = (lane >> 4) * 8;
    for (int ct = 0; ct < 2; ++ct) {
        int col0 = (wv * 2 + ct) * 16;
        int bcol = col0 + (lane & 15);
        v8f acc = {0.f, 0.f, 0.f, 0.f, 0.f, 0.f, 0.f, 0.f};
#pragma unroll
        for (int kt = 0; kt < 8; ++kt) {
            int kb = kt * 32 + koff;
            BFrag a, bq;
            a.u[0] = *(const uint4*)&As[arow][kb];
            a.u[1] = *(const uint4*)&As[arow][kb + 16];
            const unsigned short* wp = WT + (size_t)bcol * 256 + kb;
            bq.u[0] = *(const uint4*)wp;
            bq.u[1] = *(const uint4*)(wp + 16);
            acc = __builtin_amdgcn_wmma_f32_16x16x32_bf16(
                false, a.v, false, bq.v, (short)0, acc, false, false);
        }
        int rbase = (lane >> 4) * 8, cc = col0 + (lane & 15);
        float bb = bias[cc];
#pragma unroll
        for (int v = 0; v < 8; ++v)
            out[(size_t)(row0 + rbase + v) * 256 + cc] = acc[v] + bb;
    }
}

// ---------------- raw scores: S[b,h,e,m] = Qe . K / sqrt(32) ----------------
__global__ void k_scores(float* __restrict__ S, const float* __restrict__ Q,
                         const float* __restrict__ K) {
    int o = blockIdx.x * 256 + threadIdx.x;           // NB*NH*NE*NM threads
    int m = o % NM;
    int e = (o / NM) % NE;
    int h = (o / (NM * NE)) % NH;
    int b = o / (NM * NE * NH);
    const float* q = Q + ((size_t)b * NE + e) * ND + h * 32;
    const float* k = K + ((size_t)b * NM + m) * ND + h * 32;
    float s = 0.f;
#pragma unroll
    for (int d = 0; d < 32; ++d) s += q[d] * k[d];
    S[o] = s * 0.17677669529663687f;                  // 1/sqrt(32)
}

// ---------------- fused pair kernel --------------------------------------
// grid.x = NB*PAIR_TILES, block 256. For 16 pairs: sparse softmax*V -> ctx
// (bf16 LDS) -> WMMA @ WoutT -> +bias -> +residual -> LayerNorm -> out.
__global__ void k_pairs(const float* __restrict__ S, const float* __restrict__ V,
                        const float* __restrict__ ent,
                        const unsigned short* __restrict__ WT,
                        const float* __restrict__ bias,
                        const float* __restrict__ gamma,
                        const float* __restrict__ beta,
                        const int* __restrict__ mcnt, const int* __restrict__ midx,
                        float* __restrict__ out, int q_is_j) {
    __shared__ unsigned short As[16][264];
    __shared__ float Ot[16][264];
    int blk = blockIdx.x;
    int b = blk / PAIR_TILES;
    int p0 = (blk % PAIR_TILES) * 16;
    int lane = threadIdx.x & 31, wv = threadIdx.x >> 5;

    // ---- phase 1: ctx for 2 pairs per wave (lane = dim within head) ----
    for (int rr = 0; rr < 2; ++rr) {
        int r = wv * 2 + rr;
        int pp = p0 + r;
        int i = pp / NE, j = pp % NE;
        int qe = q_is_j ? j : i;      // entity providing Q and residual
        int me = q_is_j ? i : j;      // entity owning the mentions (mask)
        int cnt = mcnt[b * NE + me];
        const int* lst = midx + ((size_t)b * NE + me) * NM;
#pragma unroll
        for (int h = 0; h < NH; ++h) {
            const float* srow = S + (((size_t)b * NH + h) * NE + qe) * NM;
            float smax = -3.0e38f;
            for (int tt = 0; tt < cnt; ++tt) smax = fmaxf(smax, srow[lst[tt]]);
            float den = 0.f, acc = 0.f;
            for (int tt = 0; tt < cnt; ++tt) {
                int m = lst[tt];
                float w = __expf(srow[m] - smax);
                den += w;
                acc += w * V[((size_t)b * NM + m) * ND + h * 32 + lane];
            }
            As[r][h * 32 + lane] = f2bf(acc / den);
        }
    }
    __syncthreads();

    // ---- phase 2: out_tile = ctx @ Wout (+bias) via WMMA ----
    int arow = lane & 15, koff = (lane >> 4) * 8;
    for (int ct = 0; ct < 2; ++ct) {
        int col0 = (wv * 2 + ct) * 16;
        int bcol = col0 + (lane & 15);
        v8f acc = {0.f, 0.f, 0.f, 0.f, 0.f, 0.f, 0.f, 0.f};
#pragma unroll
        for (int kt = 0; kt < 8; ++kt) {
            int kb = kt * 32 + koff;
            BFrag a, bq;
            a.u[0] = *(const uint4*)&As[arow][kb];
            a.u[1] = *(const uint4*)&As[arow][kb + 16];
            const unsigned short* wp = WT + (size_t)bcol * 256 + kb;
            bq.u[0] = *(const uint4*)wp;
            bq.u[1] = *(const uint4*)(wp + 16);
            acc = __builtin_amdgcn_wmma_f32_16x16x32_bf16(
                false, a.v, false, bq.v, (short)0, acc, false, false);
        }
        int rbase = (lane >> 4) * 8, cc = col0 + (lane & 15);
        float bb = bias[cc];
#pragma unroll
        for (int v = 0; v < 8; ++v)
            Ot[rbase + v][cc] = acc[v] + bb;
    }
    __syncthreads();

    // ---- phase 3: residual + LayerNorm, 2 rows per wave ----
    for (int rr = 0; rr < 2; ++rr) {
        int r = wv * 2 + rr;
        int pp = p0 + r;
        int i = pp / NE, j = pp % NE;
        int qe = q_is_j ? j : i;
        const float* res = ent + ((size_t)b * NE + qe) * ND;
        float x[8], s1 = 0.f, s2 = 0.f;
#pragma unroll
        for (int tt = 0; tt < 8; ++tt) {
            int cc = lane + tt * 32;
            float v = Ot[r][cc] + res[cc];
            x[tt] = v; s1 += v; s2 += v * v;
        }
#pragma unroll
        for (int off = 16; off > 0; off >>= 1) {      // wave32 reduction
            s1 += __shfl_xor(s1, off, 32);
            s2 += __shfl_xor(s2, off, 32);
        }
        float mean = s1 * (1.f / ND);
        float var  = s2 * (1.f / ND) - mean * mean;
        float inv  = rsqrtf(var + 1e-5f);
        float* orow = out + ((size_t)b * NPAIR + pp) * ND;
#pragma unroll
        for (int tt = 0; tt < 8; ++tt) {
            int cc = lane + tt * 32;
            orow[cc] = (x[tt] - mean) * inv * gamma[cc] + beta[cc];
        }
    }
}

// ---------------- host launcher ----------------
extern "C" void kernel_launch(void* const* d_in, const int* in_sizes, int n_in,
                              void* d_out, int out_size, void* d_ws, size_t ws_size,
                              hipStream_t stream) {
    (void)in_sizes; (void)n_in; (void)out_size; (void)ws_size;
    const int*   info      = (const int*)d_in[0];
    const float* entities  = (const float*)d_in[2];
    const float* mentions  = (const float*)d_in[3];
    const float* sentences = (const float*)d_in[4];
    const float* W1 = (const float*)d_in[5];
    const float* b1 = (const float*)d_in[6];
    const float* ln1 = (const float*)d_in[7];
    const float* W2 = (const float*)d_in[8];
    const float* b2 = (const float*)d_in[9];
    const float* ln2 = (const float*)d_in[10];
    float* out = (float*)d_out;

    char* ws = (char*)d_ws;
    unsigned short* w1t    = (unsigned short*)(ws + 0);          //  4*65536 bf16
    unsigned short* w2t    = (unsigned short*)(ws + 524288);
    unsigned short* ent_bf = (unsigned short*)(ws + 1048576);    // 1536*256
    unsigned short* men_bf = (unsigned short*)(ws + 1835008);    // 3072*256
    unsigned short* msn_bf = (unsigned short*)(ws + 3407872);    // 3072*256
    float* Q1 = (float*)(ws + 4980736);                          // 1536*256
    float* Q2 = (float*)(ws + 6553600);
    float* K1 = (float*)(ws + 8126464);                          // 3072*256
    float* K2 = (float*)(ws + 11272192);
    float* V1 = (float*)(ws + 14417920);
    float* V2 = (float*)(ws + 17563648);
    float* S1 = (float*)(ws + 20709376);                         // 32*8*48*96
    float* S2 = (float*)(ws + 25427968);
    int* cnt  = (int*)(ws + 30146560);                           // 1536
    int* idx  = (int*)(ws + 30152704);                           // 1536*96

    // prep
    k_cast_transpose_w<<<1024, 256, 0, stream>>>(w1t, W1);
    k_cast_transpose_w<<<1024, 256, 0, stream>>>(w2t, W2);
    k_cast_bf<<<1536, 256, 0, stream>>>(ent_bf, entities, NB * NE * ND);
    k_cast_bf<<<3072, 256, 0, stream>>>(men_bf, mentions, NB * NM * ND);
    k_gather_sen<<<3072, 256, 0, stream>>>(msn_bf, sentences, info);
    k_build_lists<<<6, 256, 0, stream>>>(cnt, idx, info);

    // projections (Q from entities; K from men_sen; V from mentions)
    k_proj_gemm<<<96, 256, 0, stream>>>(ent_bf, w1t,            b1,       Q1);
    k_proj_gemm<<<192, 256, 0, stream>>>(msn_bf, w1t + 65536,   b1 + 256, K1);
    k_proj_gemm<<<192, 256, 0, stream>>>(men_bf, w1t + 131072,  b1 + 512, V1);
    k_proj_gemm<<<96, 256, 0, stream>>>(ent_bf, w2t,            b2,       Q2);
    k_proj_gemm<<<192, 256, 0, stream>>>(msn_bf, w2t + 65536,   b2 + 256, K2);
    k_proj_gemm<<<192, 256, 0, stream>>>(men_bf, w2t + 131072,  b2 + 512, V2);

    // raw scores
    k_scores<<<4608, 256, 0, stream>>>(S1, Q1, K1);
    k_scores<<<4608, 256, 0, stream>>>(S2, Q2, K2);

    // fused pair attention + output projection + residual + LN
    const int grid = NB * PAIR_TILES;                            // 4608
    k_pairs<<<grid, 256, 0, stream>>>(S1, V1, entities, w1t + 196608, b1 + 768,
                                      ln1, ln1 + 256, cnt, idx, out, 1);
    k_pairs<<<grid, 256, 0, stream>>>(S2, V2, entities, w2t + 196608, b2 + 768,
                                      ln2, ln2 + 256, cnt, idx,
                                      out + (size_t)NB * NPAIR * ND, 0);
}
